// VitBlock_27023934227269
// MI455X (gfx1250) — matrix operand order repaired and checked by reading, not compiled, not run on hardware
//
#include <hip/hip_runtime.h>
#include <hip/hip_bf16.h>
#include <math.h>

// ---------------- problem constants ----------------
#define DIMC   768
#define NHEAD  12
#define HDIM   64
#define HIDDEN 3072
#define SEQ    4096
#define BATCH  16
#define TOKENS (BATCH * SEQ)   // 65536
#define NBH    (BATCH * NHEAD) // 192

typedef __bf16 bf16_t;
typedef __attribute__((ext_vector_type(16))) __bf16 v16bf;
typedef __attribute__((ext_vector_type(8)))  __bf16 v8bf;
typedef __attribute__((ext_vector_type(8)))  float  v8f;

// LDS tile geometry: 128 rows x 32 K-cols bf16, padded row stride 80 B
// (bank stride 20 dwords -> 16 distinct banks for 16 consecutive rows, 16B aligned chunks)
#define ROWB      80u
#define TILE_B    (128u * ROWB)      // 10240 B per (A or B) tile
#define STAGE_B   (2u * TILE_B)      // 20480 B per stage (A + B)
#define SMEM_B    (2u * STAGE_B)     // 40960 B double buffered

__device__ __forceinline__ float elu1f(float v) {
    return v > 0.0f ? v + 1.0f : __expf(v);
}

// ================= big GEMM: LDS-staged, async-copy double buffered =================
// C[M,N] = A[M,K] * Bt[N,K]^T, f32 accum. M%128==0, N%128==0, K%32==0.
// mode 0: QKV  (+bias, elu+1 on q/k, scatter q->[B,H,N,D], k,v->[B,H,D,N] bf16)
// mode 3: PROJ (+bias + residual -> f32 out)
// mode 4: FC1  (+bias, exact GELU -> bf16)
// mode 5: FC2  (+bias, += f32 out)
__global__ __launch_bounds__(256) void gemm_bf16_lds(
    const bf16_t* __restrict__ A, const bf16_t* __restrict__ Bt,
    const float* __restrict__ bias,
    int M, int N, int K, int mode,
    float* outF, const float* __restrict__ resid,
    bf16_t* outQ, bf16_t* outKT, bf16_t* outVT, bf16_t* outBf)
{
    __shared__ __align__(16) char smem[SMEM_B];

    const int t    = threadIdx.x;
    const int lane = t & 31;
    const int wave = t >> 5;
    const int wm   = wave & 3;     // 4 waves along M, 32 rows each
    const int wn   = wave >> 2;    // 2 waves along N, 64 cols each

    const int rowBlock = blockIdx.x * 128;
    const int colBlock = blockIdx.y * 128;
    const bf16_t* aBase = A  + (size_t)rowBlock * K;
    const bf16_t* bBase = Bt + (size_t)colBlock * K;

    const unsigned smemBase = (unsigned)(uintptr_t)smem;
    const int r0 = t >> 2;         // this thread's transfer rows: r0, r0+64
    const int ch = t & 3;          // 16B chunk within the 64B row payload

    // issue one stage of async copies: 2 A + 2 B b128 transfers per thread
    auto issue = [&](int k0, int buf) {
        unsigned la = smemBase + (unsigned)buf * STAGE_B;
        unsigned lb = la + TILE_B;
#pragma unroll
        for (int j = 0; j < 2; ++j) {
            int row = r0 + j * 64;
            unsigned long long ga =
                (unsigned long long)(uintptr_t)(aBase + (size_t)row * K + k0 + ch * 8);
            unsigned da = la + (unsigned)row * ROWB + (unsigned)ch * 16u;
            asm volatile("global_load_async_to_lds_b128 %0, %1, off"
                         :: "v"(da), "v"(ga) : "memory");
            unsigned long long gb =
                (unsigned long long)(uintptr_t)(bBase + (size_t)row * K + k0 + ch * 8);
            unsigned db = lb + (unsigned)row * ROWB + (unsigned)ch * 16u;
            asm volatile("global_load_async_to_lds_b128 %0, %1, off"
                         :: "v"(db), "v"(gb) : "memory");
        }
    };

    // A fragment (16x32): lane<16 row=lane K={0..7,16..23}; lane>=16 row=lane-16 K={8..15,24..31}
    auto fragA = [&](unsigned off) -> v16bf {
        unsigned p = off + (unsigned)(lane & 15) * ROWB + (unsigned)(lane >> 4) * 16u;
        v8bf c0 = *reinterpret_cast<const v8bf*>(&smem[p]);
        v8bf c1 = *reinterpret_cast<const v8bf*>(&smem[p + 32]);
        v16bf f;
#pragma unroll
        for (int i = 0; i < 8; ++i) { f[i] = c0[i]; f[i + 8] = c1[i]; }
        return f;
    };
    // B fragment (32x16 from N-major Bt rows): lane<16 col=lane K={0..7,16..23}; lane>=16 K={8..15,24..31}
    auto fragB = [&](unsigned off) -> v16bf {
        unsigned p = off + (unsigned)(lane & 15) * ROWB + (unsigned)(lane >> 4) * 16u;
        v8bf c0 = *reinterpret_cast<const v8bf*>(&smem[p]);
        v8bf c1 = *reinterpret_cast<const v8bf*>(&smem[p + 32]);
        v16bf f;
#pragma unroll
        for (int i = 0; i < 8; ++i) { f[i] = c0[i]; f[i + 8] = c1[i]; }
        return f;
    };

    v8f acc[2][4] = {};
    const int nst = K >> 5;

    issue(0, 0);
    for (int s = 0; s < nst; ++s) {
        if (s + 1 < nst) {
            issue((s + 1) << 5, (s + 1) & 1);
            asm volatile("s_wait_asynccnt 0x4" ::: "memory");  // stage s retired
        } else {
            asm volatile("s_wait_asynccnt 0x0" ::: "memory");
        }
        __syncthreads();   // every wave's stage-s copies complete -> tile valid

        const unsigned aOff = (unsigned)(s & 1) * STAGE_B + (unsigned)(wm * 32) * ROWB;
        const unsigned bOff = (unsigned)(s & 1) * STAGE_B + TILE_B + (unsigned)(wn * 64) * ROWB;

        v16bf a0 = fragA(aOff);
        v16bf a1 = fragA(aOff + 16u * ROWB);
        v16bf b0 = fragB(bOff);
        v16bf b1 = fragB(bOff + 16u * ROWB);
        v16bf b2 = fragB(bOff + 32u * ROWB);
        v16bf b3 = fragB(bOff + 48u * ROWB);

        acc[0][0] = __builtin_amdgcn_wmma_f32_16x16x32_bf16(false, a0, false, b0, (short)0, acc[0][0], false, false);
        acc[0][1] = __builtin_amdgcn_wmma_f32_16x16x32_bf16(false, a0, false, b1, (short)0, acc[0][1], false, false);
        acc[0][2] = __builtin_amdgcn_wmma_f32_16x16x32_bf16(false, a0, false, b2, (short)0, acc[0][2], false, false);
        acc[0][3] = __builtin_amdgcn_wmma_f32_16x16x32_bf16(false, a0, false, b3, (short)0, acc[0][3], false, false);
        acc[1][0] = __builtin_amdgcn_wmma_f32_16x16x32_bf16(false, a1, false, b0, (short)0, acc[1][0], false, false);
        acc[1][1] = __builtin_amdgcn_wmma_f32_16x16x32_bf16(false, a1, false, b1, (short)0, acc[1][1], false, false);
        acc[1][2] = __builtin_amdgcn_wmma_f32_16x16x32_bf16(false, a1, false, b2, (short)0, acc[1][2], false, false);
        acc[1][3] = __builtin_amdgcn_wmma_f32_16x16x32_bf16(false, a1, false, b3, (short)0, acc[1][3], false, false);

        __syncthreads();   // all reads of buf[s&1] done before it is refilled
    }

    // C/D layout: VGPR i -> row (lane>=16?8:0)+i, col lane&15 within each 16x16 tile
    const int cRow0 = (lane >> 4) * 8;
    const int cCol  = lane & 15;
    const int rowBase = rowBlock + wm * 32;
    const int colBase = colBlock + wn * 64;

#pragma unroll
    for (int tm = 0; tm < 2; ++tm) {
#pragma unroll
        for (int tn = 0; tn < 4; ++tn) {
#pragma unroll
            for (int i = 0; i < 8; ++i) {
                const int r = rowBase + tm * 16 + cRow0 + i;
                const int c = colBase + tn * 16 + cCol;
                float val = acc[tm][tn][i];

                if (mode == 0) {
                    val += bias[c];
                    int which = c / DIMC;
                    int rem   = c - which * DIMC;
                    int h = rem >> 6, d = rem & 63;
                    int b = r >> 12, sTok = r & 4095;
                    size_t hb = (size_t)b * NHEAD + h;
                    if (which == 0) {
                        outQ[(hb * SEQ + sTok) * HDIM + d] = (bf16_t)elu1f(val);
                    } else if (which == 1) {
                        outKT[(hb * HDIM + d) * SEQ + sTok] = (bf16_t)elu1f(val);
                    } else {
                        outVT[(hb * HDIM + d) * SEQ + sTok] = (bf16_t)val;
                    }
                } else if (mode == 3) {
                    size_t idx = (size_t)r * N + c;
                    outF[idx] = val + bias[c] + resid[idx];
                } else if (mode == 4) {
                    float v = val + bias[c];
                    float g = 0.5f * v * (1.0f + erff(v * 0.70710678118654752f));
                    outBf[(size_t)r * N + c] = (bf16_t)g;
                } else { // mode 5
                    size_t idx = (size_t)r * N + c;
                    outF[idx] = outF[idx] + val + bias[c];
                }
            }
        }
    }
}

// ================= small per-head GEMMs: direct-global WMMA (no barriers) =================
// mode 1: KV  per head: kT[64,4096] x vT[64,4096]^T -> kv^T [e][d] bf16
// mode 2: OUT per head: q[4096,64] x kvT[64,64]^T -> *z[s] -> attn [B,N,C] bf16
__device__ __forceinline__ v16bf gload_fragA(const bf16_t* __restrict__ base, int ld, int lane) {
    int row  = lane & 15;
    int half = lane >> 4;
    const bf16_t* p = base + (size_t)row * ld + half * 8;
    v8bf c0 = *reinterpret_cast<const v8bf*>(p);
    v8bf c1 = *reinterpret_cast<const v8bf*>(p + 16);
    v16bf f;
#pragma unroll
    for (int i = 0; i < 8; ++i) { f[i] = c0[i]; f[i + 8] = c1[i]; }
    return f;
}
__device__ __forceinline__ v16bf gload_fragB(const bf16_t* __restrict__ baseT, int ld, int lane) {
    int col  = lane & 15;
    int half = lane >> 4;
    const bf16_t* p = baseT + (size_t)col * ld + half * 8;
    v8bf c0 = *reinterpret_cast<const v8bf*>(p);
    v8bf c1 = *reinterpret_cast<const v8bf*>(p + 16);
    v16bf f;
#pragma unroll
    for (int i = 0; i < 8; ++i) { f[i] = c0[i]; f[i + 8] = c1[i]; }
    return f;
}

__global__ __launch_bounds__(256) void gemm_bf16_direct(
    const bf16_t* __restrict__ A, const bf16_t* __restrict__ Bt,
    int M, int N, int K, int mode,
    const float* __restrict__ zscale, bf16_t* outBf)
{
    const int lane = threadIdx.x & 31;
    const int wave = threadIdx.x >> 5;
    const int wm = wave & 3;
    const int wn = wave >> 2;
    const int bh = blockIdx.z;

    if (mode == 1) {
        A  += (size_t)bh * HDIM * SEQ;
        Bt += (size_t)bh * HDIM * SEQ;
    } else { // mode 2
        A      += (size_t)bh * SEQ * HDIM;
        Bt     += (size_t)bh * HDIM * HDIM;
        zscale += (size_t)bh * SEQ;
    }

    const int rowBase = blockIdx.x * 128 + wm * 32;
    const int colBase = blockIdx.y * 64  + wn * 32;
    if (rowBase >= M || colBase >= N) return;   // whole-wave guard (no barriers)

    v8f acc[2][2] = {};
    const bf16_t* aP = A  + (size_t)rowBase * K;
    const bf16_t* bP = Bt + (size_t)colBase * K;

    for (int k0 = 0; k0 < K; k0 += 32) {
        v16bf a0 = gload_fragA(aP + k0, K, lane);
        v16bf a1 = gload_fragA(aP + (size_t)16 * K + k0, K, lane);
        v16bf b0 = gload_fragB(bP + k0, K, lane);
        v16bf b1 = gload_fragB(bP + (size_t)16 * K + k0, K, lane);
        if (k0 + 256 <= K) {
            __builtin_prefetch((const void*)(aP + (size_t)lane * K + k0 + 256), 0, 3);
            __builtin_prefetch((const void*)(bP + (size_t)lane * K + k0 + 256), 0, 3);
        }
        acc[0][0] = __builtin_amdgcn_wmma_f32_16x16x32_bf16(false, a0, false, b0, (short)0, acc[0][0], false, false);
        acc[0][1] = __builtin_amdgcn_wmma_f32_16x16x32_bf16(false, a0, false, b1, (short)0, acc[0][1], false, false);
        acc[1][0] = __builtin_amdgcn_wmma_f32_16x16x32_bf16(false, a1, false, b0, (short)0, acc[1][0], false, false);
        acc[1][1] = __builtin_amdgcn_wmma_f32_16x16x32_bf16(false, a1, false, b1, (short)0, acc[1][1], false, false);
    }

    const int cRow0 = (lane >> 4) * 8;
    const int cCol  = lane & 15;
#pragma unroll
    for (int tm = 0; tm < 2; ++tm) {
#pragma unroll
        for (int tn = 0; tn < 2; ++tn) {
#pragma unroll
            for (int i = 0; i < 8; ++i) {
                const int r = rowBase + tm * 16 + cRow0 + i;
                const int c = colBase + tn * 16 + cCol;
                float val = acc[tm][tn][i];
                if (mode == 1) {
                    // kv[d=r][e=c] -> kvT[e][d]
                    outBf[(size_t)bh * HDIM * HDIM + (size_t)c * HDIM + r] = (bf16_t)val;
                } else {
                    float v = val * zscale[r];
                    int b = bh / NHEAD, h = bh - b * NHEAD;
                    outBf[((size_t)b * SEQ + r) * DIMC + h * HDIM + c] = (bf16_t)v;
                }
            }
        }
    }
}

// ---------------- LayerNorm (f32 in, bf16 out), one token per 256-thread block ----
__global__ __launch_bounds__(256) void ln_bf16_kernel(
    const float* __restrict__ x, const float* __restrict__ w,
    const float* __restrict__ b, bf16_t* __restrict__ out)
{
    __shared__ float red[8];
    const int t = threadIdx.x;
    const size_t base = (size_t)blockIdx.x * DIMC;

    float v[3];
    float s = 0.0f;
#pragma unroll
    for (int i = 0; i < 3; ++i) { v[i] = x[base + t + i * 256]; s += v[i]; }
#pragma unroll
    for (int off = 16; off > 0; off >>= 1) s += __shfl_down(s, off, 32);
    if ((t & 31) == 0) red[t >> 5] = s;
    __syncthreads();
    float tot = 0.0f;
#pragma unroll
    for (int i = 0; i < 8; ++i) tot += red[i];
    const float mean = tot * (1.0f / (float)DIMC);

    float vs = 0.0f;
#pragma unroll
    for (int i = 0; i < 3; ++i) { float d = v[i] - mean; vs += d * d; }
#pragma unroll
    for (int off = 16; off > 0; off >>= 1) vs += __shfl_down(vs, off, 32);
    __syncthreads();
    if ((t & 31) == 0) red[t >> 5] = vs;
    __syncthreads();
    float vtot = 0.0f;
#pragma unroll
    for (int i = 0; i < 8; ++i) vtot += red[i];
    const float inv = rsqrtf(vtot * (1.0f / (float)DIMC) + 1e-8f);

#pragma unroll
    for (int i = 0; i < 3; ++i) {
        int c = t + i * 256;
        out[base + c] = (bf16_t)((v[i] - mean) * inv * w[c] + b[c]);
    }
}

// ---------------- f32 [K,N] -> bf16 transposed [N,K] weight prep ----------------
__global__ __launch_bounds__(256) void wtrans_kernel(
    const float* __restrict__ W, bf16_t* __restrict__ Wt, int K, int N)
{
    size_t gid = (size_t)blockIdx.x * 256 + threadIdx.x;
    if (gid >= (size_t)K * N) return;
    size_t k = gid / N;
    size_t n = gid - k * N;
    Wt[n * K + k] = (bf16_t)W[gid];
}

// ---------------- ksum[b,h,d] = sum_s kT[b,h,d,s] ----------------
__global__ __launch_bounds__(64) void ksum_kernel(
    const bf16_t* __restrict__ kT, float* __restrict__ ksum)
{
    const int bh = blockIdx.x;
    const int d  = threadIdx.x;
    const bf16_t* p = kT + ((size_t)bh * HDIM + d) * SEQ;
    float s = 0.0f;
    for (int i = 0; i < SEQ; i += 8) {
        v8bf c = *reinterpret_cast<const v8bf*>(p + i);
#pragma unroll
        for (int j = 0; j < 8; ++j) s += (float)c[j];
    }
    ksum[bh * HDIM + d] = s;
}

// ---------------- z[b,h,s] = 1 / (q[s,:] . ksum[b,h,:]) ----------------
__global__ __launch_bounds__(256) void zinv_kernel(
    const bf16_t* __restrict__ q, const float* __restrict__ ksum, float* __restrict__ z)
{
    size_t gid = (size_t)blockIdx.x * 256 + threadIdx.x;   // over NBH*SEQ
    int bh = (int)(gid >> 12);
    const bf16_t* qp = q + gid * HDIM;
    const float*  kp = ksum + (size_t)bh * HDIM;
    float acc = 0.0f;
#pragma unroll
    for (int d = 0; d < HDIM; ++d) acc += (float)qp[d] * kp[d];
    z[gid] = 1.0f / acc;
}

// ---------------- host launch ----------------
extern "C" void kernel_launch(void* const* d_in, const int* in_sizes, int n_in,
                              void* d_out, int out_size, void* d_ws, size_t ws_size,
                              hipStream_t stream) {
    const float* x      = (const float*)d_in[0];
    const float* qkv_w  = (const float*)d_in[1];
    const float* qkv_b  = (const float*)d_in[2];
    const float* proj_w = (const float*)d_in[3];
    const float* proj_b = (const float*)d_in[4];
    const float* fc1_w  = (const float*)d_in[5];
    const float* fc1_b  = (const float*)d_in[6];
    const float* fc2_w  = (const float*)d_in[7];
    const float* fc2_b  = (const float*)d_in[8];
    const float* ln1_w  = (const float*)d_in[9];
    const float* ln1_b  = (const float*)d_in[10];
    const float* ln2_w  = (const float*)d_in[11];
    const float* ln2_b  = (const float*)d_in[12];
    float* out = (float*)d_out;

    char* ws = (char*)d_ws;
    size_t off = 0;
    auto alloc = [&](size_t bytes) -> char* {
        char* p = ws + off;
        off += (bytes + 255) & ~(size_t)255;
        return p;
    };
    const size_t actB = (size_t)TOKENS * DIMC * sizeof(bf16_t);   // 96 MiB, 256-aligned

    bf16_t* qkvWt = (bf16_t*)alloc((size_t)2304 * 768 * sizeof(bf16_t));
    bf16_t* projWt= (bf16_t*)alloc((size_t)768  * 768 * sizeof(bf16_t));
    bf16_t* fc1Wt = (bf16_t*)alloc((size_t)3072 * 768 * sizeof(bf16_t));
    bf16_t* fc2Wt = (bf16_t*)alloc((size_t)768 * 3072 * sizeof(bf16_t));
    bf16_t* xn    = (bf16_t*)alloc(actB);   // LN1 out; reused as attn out / gelu span head
    bf16_t* qbuf  = (bf16_t*)alloc(actB);   // [B,H,N,D]
    bf16_t* kT    = (bf16_t*)alloc(actB);   // [B,H,D,N]
    bf16_t* vT    = (bf16_t*)alloc(actB);   // [B,H,D,N]
    bf16_t* hbuf  = (bf16_t*)alloc(actB);   // LN2 out
    bf16_t* kvT   = (bf16_t*)alloc((size_t)NBH * HDIM * HDIM * sizeof(bf16_t));
    float*  ksum  = (float*) alloc((size_t)NBH * HDIM * sizeof(float));
    float*  zbuf  = (float*) alloc((size_t)NBH * SEQ * sizeof(float));
    bf16_t* attn  = xn;                     // xn dead after QKV GEMM
    bf16_t* gelu  = xn;                     // spans xn..vT (= TOKENS*HIDDEN bf16), all dead by FC1

    dim3 blk(256);

    // 1) bf16 transposed weights
    wtrans_kernel<<<dim3((768u * 2304u + 255u) / 256u), blk, 0, stream>>>(qkv_w, qkvWt, 768, 2304);
    wtrans_kernel<<<dim3((768u * 768u  + 255u) / 256u), blk, 0, stream>>>(proj_w, projWt, 768, 768);
    wtrans_kernel<<<dim3((768u * 3072u + 255u) / 256u), blk, 0, stream>>>(fc1_w, fc1Wt, 768, 3072);
    wtrans_kernel<<<dim3((3072u * 768u + 255u) / 256u), blk, 0, stream>>>(fc2_w, fc2Wt, 3072, 768);

    // 2) LN1 -> bf16
    ln_bf16_kernel<<<dim3(TOKENS), blk, 0, stream>>>(x, ln1_w, ln1_b, xn);

    // 3) QKV GEMM (+bias, elu+1, scatter)  [M=65536, N=2304, K=768]
    gemm_bf16_lds<<<dim3(TOKENS / 128, 2304 / 128, 1), blk, 0, stream>>>(
        xn, qkvWt, qkv_b, TOKENS, 2304, 768, 0,
        nullptr, nullptr, qbuf, kT, vT, nullptr);

    // 4) ksum, kv = kT x vT per head, z = 1/(q . ksum)
    ksum_kernel<<<dim3(NBH), dim3(64), 0, stream>>>(kT, ksum);
    gemm_bf16_direct<<<dim3(1, 1, NBH), blk, 0, stream>>>(
        kT, vT, HDIM, HDIM, SEQ, 1, nullptr, kvT);
    zinv_kernel<<<dim3(NBH * SEQ / 256), blk, 0, stream>>>(qbuf, ksum, zbuf);

    // 5) out = (q x kv) * z, gathered to [B,N,C]
    gemm_bf16_direct<<<dim3(SEQ / 128, 1, NBH), blk, 0, stream>>>(
        qbuf, kvT, SEQ, HDIM, HDIM, 2, zbuf, attn);

    // 6) proj (+bias + residual x) -> d_out  [M=65536, N=768, K=768]
    gemm_bf16_lds<<<dim3(TOKENS / 128, DIMC / 128, 1), blk, 0, stream>>>(
        attn, projWt, proj_b, TOKENS, DIMC, DIMC, 3,
        out, x, nullptr, nullptr, nullptr, nullptr);

    // 7) LN2 -> bf16
    ln_bf16_kernel<<<dim3(TOKENS), blk, 0, stream>>>(out, ln2_w, ln2_b, hbuf);

    // 8) fc1 (+bias, exact GELU) -> bf16  [M=65536, N=3072, K=768]
    gemm_bf16_lds<<<dim3(TOKENS / 128, HIDDEN / 128, 1), blk, 0, stream>>>(
        hbuf, fc1Wt, fc1_b, TOKENS, HIDDEN, DIMC, 4,
        nullptr, nullptr, nullptr, nullptr, nullptr, gelu);

    // 9) fc2 (+bias, += d_out)  [M=65536, N=768, K=3072]
    gemm_bf16_lds<<<dim3(TOKENS / 128, DIMC / 128, 1), blk, 0, stream>>>(
        gelu, fc2Wt, fc2_b, TOKENS, DIMC, HIDDEN, 5,
        out, nullptr, nullptr, nullptr, nullptr, nullptr);

    (void)in_sizes; (void)n_in; (void)out_size; (void)ws_size;
}
